// WeekEncoder_57415122812929
// MI455X (gfx1250) — matrix operand order repaired and checked by reading, not compiled, not run on hardware
//
#include <hip/hip_runtime.h>

typedef __attribute__((ext_vector_type(16))) _Float16 v16h;
typedef __attribute__((ext_vector_type(8)))  float    v8f;

#define HDIM 256
#define G4   1024   // 4*H gate columns

union Frag16 { v16h v; float4 f4[2]; };

__device__ __forceinline__ float sigmoidf_(float x) {
    return 1.0f / (1.0f + __expf(-x));
}

// ---------------------------------------------------------------------------
// Pack [Wih | zero-pad | Whh] rows into one f16 weight matrix [1024][K]
// so gates = [x_pad | h] @ Wpack^T in a single K loop.
// ---------------------------------------------------------------------------
__global__ __launch_bounds__(256) void pack_weights_kernel(
    const float* __restrict__ Wih, const float* __restrict__ Whh,
    _Float16* __restrict__ dst, int in_dim, int KX, int K)
{
    int idx = blockIdx.x * blockDim.x + threadIdx.x;
    if (idx >= G4 * K) return;
    int j = idx / K, c = idx - j * K;
    float v;
    if (c < in_dim)      v = Wih[j * in_dim + c];
    else if (c < KX)     v = 0.0f;
    else                 v = Whh[j * HDIM + (c - KX)];
    dst[idx] = (_Float16)v;
}

// ---------------------------------------------------------------------------
// Build X0[n][t][64] f16 = [emb[id](32) | feats(7) | zero(25)]
// x is [B=64][Wk=16][T=64][8] fp32, ids in channel 0.
// ---------------------------------------------------------------------------
__global__ __launch_bounds__(256) void embed_kernel(
    const float* __restrict__ x, const float* __restrict__ emb,
    _Float16* __restrict__ X0)
{
    int nt = blockIdx.x * blockDim.x + threadIdx.x;   // over N*T = 65536
    if (nt >= 1024 * 64) return;
    const float* xr = x + (size_t)nt * 8;
    int id = (int)xr[0];
    const float* er = emb + (size_t)id * 32;
    _Float16* dst = X0 + (size_t)nt * 64;
    #pragma unroll
    for (int i = 0; i < 32; ++i) dst[i] = (_Float16)er[i];
    #pragma unroll
    for (int i = 0; i < 7; ++i)  dst[32 + i] = (_Float16)xr[1 + i];
    #pragma unroll
    for (int i = 0; i < 25; ++i) dst[39 + i] = (_Float16)0.0f;
}

// lengths[n] = max(sum_t mask[n][t], 1)   (mask is a 1-byte bool array)
__global__ __launch_bounds__(256) void lengths_kernel(
    const unsigned char* __restrict__ mask, int* __restrict__ lengths)
{
    int n = blockIdx.x * blockDim.x + threadIdx.x;
    if (n >= 1024) return;
    int s = 0;
    #pragma unroll
    for (int t = 0; t < 64; ++t) s += mask[n * 64 + t] ? 1 : 0;
    lengths[n] = (s < 1) ? 1 : s;
}

// ---------------------------------------------------------------------------
// One LSTM layer. Each workgroup owns 16 sequences and iterates all T steps
// locally (recurrence is batch-independent -> no grid sync needed).
// 8 waves x 8 tiles of 16 gate columns = 1024 columns; v_wmma_f32_16x16x32_f16
// over fused K = KX (x part, padded) + 256 (h part).
//   A (16 x K, f16)   : LDS, rows padded by 8 halves to break bank conflicts
//   B (K x 16, f16)   : streamed from packed weights via L2 every step
//   gates (16 x 1024) : fp32 LDS scratch for the cross-wave gate exchange
//   c state (16 x 256): fp32 LDS
// ---------------------------------------------------------------------------
template <int KX, int KTOT, bool HAS_MASK>
__global__ __launch_bounds__(256) void lstm_layer_kernel(
    const _Float16* __restrict__ X,        // [nSeq][T][KX]
    const _Float16* __restrict__ Wpack,    // [1024][KTOT]
    const float*    __restrict__ bias,     // [1024]
    const int*      __restrict__ lengths,  // [nSeq] (HAS_MASK only)
    int T,
    _Float16* __restrict__ out_seq,        // [nSeq][T][256] or null
    _Float16* __restrict__ out_final_h16,  // [nSeq][256] or null
    float*    __restrict__ out_final_h32,  // [nSeq][256] or null
    float*    __restrict__ out_final_c32)  // [nSeq][256] or null
{
    constexpr int ASTR = KTOT + 8;         // padded A row stride (halves)

    extern __shared__ char smem_raw[];
    float*    gates = (float*)smem_raw;            // 16*1024 fp32 = 64 KB
    float*    Cs    = gates + 16 * G4;             // 16*256  fp32 = 16 KB
    _Float16* A     = (_Float16*)(Cs + 16 * HDIM); // 16*ASTR f16
    int*      slen  = (int*)(A + 16 * ASTR);

    const int tid  = threadIdx.x;
    const int wave = tid >> 5;
    const int lane = tid & 31;
    const int lhi  = lane >> 4;    // half-wave select
    const int lmod = lane & 15;
    const int seq_base = blockIdx.x * 16;

    // Per-lane loop-invariant base pointers: tile/k offsets become immediates.
    const _Float16* wbase = Wpack + (size_t)(wave * 128 + lmod) * KTOT + lhi * 8;
    const _Float16* abase = A + lmod * ASTR + lhi * 8;

    // Bias columns held in registers across the whole timestep loop.
    float bcol[8];
    #pragma unroll
    for (int tt = 0; tt < 8; ++tt)
        bcol[tt] = bias[(wave * 8 + tt) * 16 + lmod];

    // zero h region of A and the c state
    for (int idx = tid; idx < 16 * HDIM; idx += 256) {
        int r = idx >> 8, c = idx & (HDIM - 1);
        A[r * ASTR + KX + c] = (_Float16)0.0f;
        Cs[idx] = 0.0f;
    }
    if (HAS_MASK && tid < 16) slen[tid] = lengths[seq_base + tid];

    const int er  = tid >> 4;      // elementwise row 0..15
    const int ec0 = tid & 15;

    #pragma clang loop unroll(disable)
    for (int t = 0; t < T; ++t) {
        // Optimization barrier: stop LICM from hoisting the (t-invariant)
        // weight loads out of the timestep loop (that path register-spills
        // to scratch). Re-reading Wpack each step hits L2 (~1 MB resident).
        asm volatile("" ::: "memory");

        // ---- stage x_t into the x region of A (16-byte vector copies) ----
        for (int idx = tid; idx < 2 * KX; idx += 256) {
            int r  = idx / (KX / 8);
            int cc = idx - r * (KX / 8);
            *(float4*)&A[r * ASTR + cc * 8] =
                *(const float4*)&X[((size_t)(seq_base + r) * T + t) * KX + cc * 8];
        }
        __syncthreads();

        // ---- gates = [x|h] @ Wpack^T + b via WMMA ----
        v8f acc[8];
        #pragma unroll
        for (int tt = 0; tt < 8; ++tt) {
            #pragma unroll
            for (int i = 0; i < 8; ++i) acc[tt][i] = bcol[tt];
        }
        #pragma clang loop unroll(disable)
        for (int k0 = 0; k0 < KTOT; k0 += 32) {
            // A fragment (16x32 f16): lane<16 -> K {k0..k0+7, k0+16..23},
            //                          lane>=16 -> K {k0+8..15, k0+24..31}
            const _Float16* ap = abase + k0;
            Frag16 af;
            af.f4[0] = *(const float4*)(ap);
            af.f4[1] = *(const float4*)(ap + 16);

            // Issue ALL 8 B-fragment loads first (one clause, immediate
            // offsets), then the 8 WMMAs -> staggered partial loadcnt waits
            // overlap L2 latency with the matrix pipe.
            const _Float16* wk = wbase + k0;
            Frag16 bf[8];
            #pragma unroll
            for (int tt = 0; tt < 8; ++tt) {
                bf[tt].f4[0] = *(const float4*)(wk + tt * 16 * KTOT);
                bf[tt].f4[1] = *(const float4*)(wk + tt * 16 * KTOT + 16);
            }
            #pragma unroll
            for (int tt = 0; tt < 8; ++tt) {
                acc[tt] = __builtin_amdgcn_wmma_f32_16x16x32_f16(
                    false, af.v, false, bf[tt].v, (short)0, acc[tt], false, false);
            }
        }
        // C layout: VGPR v -> M = v + 8*lhi, lane -> N = lmod
        #pragma unroll
        for (int tt = 0; tt < 8; ++tt) {
            int col = (wave * 8 + tt) * 16 + lmod;
            #pragma unroll
            for (int vv = 0; vv < 8; ++vv)
                gates[(vv + lhi * 8) * G4 + col] = acc[tt][vv];
        }
        __syncthreads();

        // ---- fused LSTM cell update (freeze padded steps) ----
        const bool active = HAS_MASK ? (t < slen[er]) : true;
        #pragma unroll 4
        for (int kk = 0; kk < 16; ++kk) {
            int c = ec0 + kk * 16;
            float gi = gates[er * G4 + c];
            float gf = gates[er * G4 + HDIM + c];
            float gg = gates[er * G4 + 2 * HDIM + c];
            float go = gates[er * G4 + 3 * HDIM + c];
            float cp = Cs[er * HDIM + c];
            float cn = sigmoidf_(gf) * cp + sigmoidf_(gi) * tanhf(gg);
            float hn = sigmoidf_(go) * tanhf(cn);
            if (active) {
                Cs[er * HDIM + c] = cn;
                A[er * ASTR + KX + c] = (_Float16)hn;
            }
            if (out_seq) {
                out_seq[((size_t)(seq_base + er) * T + t) * HDIM + c] =
                    A[er * ASTR + KX + c];
            }
        }
        // no barrier needed here: next x-stage touches a disjoint LDS region;
        // the top-of-loop barrier orders h writes vs. the next matmul.
    }

    __syncthreads();
    for (int idx = tid; idx < 16 * HDIM; idx += 256) {
        int r = idx >> 8, c = idx & (HDIM - 1);
        float hv = (float)A[r * ASTR + KX + c];
        if (out_final_h16) out_final_h16[(size_t)(seq_base + r) * HDIM + c] = (_Float16)hv;
        if (out_final_h32) out_final_h32[(size_t)(seq_base + r) * HDIM + c] = hv;
        if (out_final_c32) out_final_c32[(size_t)(seq_base + r) * HDIM + c] = Cs[idx];
    }
}

// ---------------------------------------------------------------------------
extern "C" void kernel_launch(void* const* d_in, const int* in_sizes, int n_in,
                              void* d_out, int out_size, void* d_ws, size_t ws_size,
                              hipStream_t stream)
{
    const float* x            = (const float*)d_in[0];
    const unsigned char* mask = (const unsigned char*)d_in[1];
    const float* emb          = (const float*)d_in[2];
    const float* t_Wih0       = (const float*)d_in[3];
    const float* t_Whh0       = (const float*)d_in[4];
    const float* t_b0         = (const float*)d_in[5];
    const float* t_Wih1       = (const float*)d_in[6];
    const float* t_Whh1       = (const float*)d_in[7];
    const float* t_b1         = (const float*)d_in[8];
    const float* w_Wih0       = (const float*)d_in[9];
    const float* w_Whh0       = (const float*)d_in[10];
    const float* w_b0         = (const float*)d_in[11];
    const float* w_Wih1       = (const float*)d_in[12];
    const float* w_Whh1       = (const float*)d_in[13];
    const float* w_b1         = (const float*)d_in[14];
    float* out = (float*)d_out;   // [wh0 | wh1 | wc0 | wc1], each 64*256 fp32

    char* ws = (char*)d_ws;
    size_t off = 0;
    auto take = [&](size_t bytes) -> char* {
        char* p = ws + off;
        off += (bytes + 255) & ~(size_t)255;
        return p;
    };
    _Float16* X0   = (_Float16*)take((size_t)1024 * 64 * 64 * 2);   // 8 MB
    _Float16* out0 = (_Float16*)take((size_t)1024 * 64 * 256 * 2);  // 32 MB
    _Float16* wseq = (_Float16*)take((size_t)1024 * 256 * 2);
    _Float16* ow0  = (_Float16*)take((size_t)1024 * 256 * 2);
    _Float16* W0t  = (_Float16*)take((size_t)1024 * 320 * 2);
    _Float16* W1t  = (_Float16*)take((size_t)1024 * 512 * 2);
    _Float16* W0w  = (_Float16*)take((size_t)1024 * 512 * 2);
    _Float16* W1w  = (_Float16*)take((size_t)1024 * 512 * 2);
    int*      lens = (int*)take(1024 * 4);

    pack_weights_kernel<<<(1024 * 320 + 255) / 256, 256, 0, stream>>>(t_Wih0, t_Whh0, W0t, 39, 64, 320);
    pack_weights_kernel<<<(1024 * 512 + 255) / 256, 256, 0, stream>>>(t_Wih1, t_Whh1, W1t, 256, 256, 512);
    pack_weights_kernel<<<(1024 * 512 + 255) / 256, 256, 0, stream>>>(w_Wih0, w_Whh0, W0w, 256, 256, 512);
    pack_weights_kernel<<<(1024 * 512 + 255) / 256, 256, 0, stream>>>(w_Wih1, w_Whh1, W1w, 256, 256, 512);
    embed_kernel<<<256, 256, 0, stream>>>(x, emb, X0);
    lengths_kernel<<<4, 256, 0, stream>>>(mask, lens);

    const size_t smem0 = 16 * 1024 * 4 + 16 * 256 * 4 + 16 * (320 + 8) * 2 + 64;
    const size_t smem1 = 16 * 1024 * 4 + 16 * 256 * 4 + 16 * (512 + 8) * 2 + 64;

    // Task-level LSTM: 1024 sequences, T=64, masked
    lstm_layer_kernel<64, 320, true><<<64, 256, smem0, stream>>>(
        X0, W0t, t_b0, lens, 64, out0, nullptr, nullptr, nullptr);
    lstm_layer_kernel<256, 512, true><<<64, 256, smem1, stream>>>(
        out0, W1t, t_b1, lens, 64, nullptr, wseq, nullptr, nullptr);

    // Week-level LSTM: 64 sequences, T=16, no mask
    lstm_layer_kernel<256, 512, false><<<4, 256, smem1, stream>>>(
        wseq, W0w, w_b0, nullptr, 16, ow0, nullptr, out + 0, out + 2 * 16384);
    lstm_layer_kernel<256, 512, false><<<4, 256, smem1, stream>>>(
        ow0, W1w, w_b1, nullptr, 16, nullptr, nullptr, out + 16384, out + 3 * 16384);
}
